// DCRNN_Down_65609920414487
// MI455X (gfx1250) — matrix operand order repaired and checked by reading, not compiled, not run on hardware
//
#include <hip/hip_runtime.h>
#include <hip/hip_bf16.h>
#include <math.h>

// ---------------------------------------------------------------------------
// MI455X (gfx1250) implementation notes:
//  * wave32, WMMA 16x16x32 bf16 -> f32 accum for all five implicit GEMMs
//  * software-pipelined K loop: TDM (tensor_load_to_lds) prefetches the next
//    weight tile into a ping-pong LDS buffer while WMMAs consume the current
//    one; one workgroup barrier per K-step
//  * deformable bilinear sampling fused into WMMA B-operand LDS staging
//  * gates conv (256x1152 GEMM, 78% of FLOPs) fused with the LSTM pointwise
//    epilogue through LDS -> the 67 MB gates tensor never touches HBM
//  * intermediates that feed GEMMs kept in bf16 (halves HBM traffic)
// ---------------------------------------------------------------------------

typedef __bf16 bf16_t;
typedef bf16_t bfx16 __attribute__((ext_vector_type(16)));
typedef bf16_t bfx8  __attribute__((ext_vector_type(8)));
typedef float  fx8   __attribute__((ext_vector_type(8)));
typedef unsigned int u32x4 __attribute__((ext_vector_type(4)));
typedef int i32x4 __attribute__((ext_vector_type(4)));
typedef int i32x8 __attribute__((ext_vector_type(8)));

#define HP 128
#define WP 128
#define HWSZ (HP * WP)
#define BATCH 4
#define KC 32          // K chunk per WMMA step
#define TILE_PIX 64    // pixels (N) per workgroup

#if defined(__has_builtin)
#if __has_builtin(__builtin_amdgcn_tensor_load_to_lds) && \
    __has_builtin(__builtin_amdgcn_s_wait_tensorcnt)
#define HAVE_TDM 1
#endif
#endif
#ifndef HAVE_TDM
#define HAVE_TDM 0
#endif

// ---------------------- Tensor Data Mover (TDM) helper ---------------------
// 2-D tile load: tile_y rows x tile_x bf16 elements, row stride = stride_x
// elements; rows >= tensor_y are zero-filled by hardware OOB handling.
// D# packing per CDNA5 ISA ch.8.3/8.4. Issued by wave 0 only; completion
// tracked with TENSORcnt and published with a workgroup barrier.
__device__ __forceinline__ void tdm_load_2d_bf16(const bf16_t* gsrc,
                                                 unsigned lds_off,
                                                 unsigned tile_x,
                                                 unsigned tile_y,
                                                 unsigned tensor_y,
                                                 unsigned stride_x) {
#if HAVE_TDM
  unsigned long long ga = (unsigned long long)(__UINTPTR_TYPE__)gsrc;
  u32x4 g0;
  g0[0] = 1u;                                    // count=1 (valid user D#)
  g0[1] = lds_off;                               // lds_addr
  g0[2] = (unsigned)(ga & 0xffffffffull);        // global_addr[31:0]
  g0[3] = (unsigned)((ga >> 32) & 0x01ffffffull) | (2u << 30);  // [56:32]|type=2
  i32x8 g1;
  unsigned tensor_x = tile_x;                    // K chunk always in-range
  g1[0] = (int)(1u << 16);                       // data_size=2B; no mcast/pad
  g1[1] = (int)((tensor_x & 0xffffu) << 16);     // tensor_dim0[15:0]
  g1[2] = (int)((tensor_x >> 16) | ((tensor_y & 0xffffu) << 16));
  g1[3] = (int)((tensor_y >> 16) | ((tile_x & 0xffffu) << 16));
  g1[4] = (int)(tile_y & 0xffffu);               // tile_dim1 (tile_dim2=0)
  g1[5] = (int)stride_x;                         // tensor_dim0_stride[31:0]
  g1[6] = 0;                                     // stride hi + dim1_stride lo
  g1[7] = 0;
  i32x4 z4 = {};
#if defined(__clang_major__) && (__clang_major__ >= 23)
  i32x8 z8 = {};
  __builtin_amdgcn_tensor_load_to_lds(g0, g1, z4, z4, z8, 0);
#else
  __builtin_amdgcn_tensor_load_to_lds(g0, g1, z4, z4, 0);
#endif
#else
  (void)gsrc; (void)lds_off; (void)tile_x; (void)tile_y; (void)tensor_y;
  (void)stride_x;
#endif
}

__device__ __forceinline__ void tdm_wait() {
#if HAVE_TDM
  __builtin_amdgcn_s_wait_tensorcnt(0);
#endif
}

// ------------------------- elementwise helpers -----------------------------

__global__ __launch_bounds__(256) void pack_bf16_kernel(
    const float* __restrict__ src, bf16_t* __restrict__ dst, int n) {
  int i = blockIdx.x * 256 + threadIdx.x;
  if (i < n) dst[i] = (bf16_t)src[i];
}

// maxpool 2x2 on (4,32,256,256) -> bf16 (4,32,128,128)
__global__ __launch_bounds__(256) void maxpool_bf_kernel(
    const float* __restrict__ x, bf16_t* __restrict__ xp) {
  size_t idx = (size_t)blockIdx.x * 256 + threadIdx.x;  // over 4*32*128*128
  int wo = idx % WP;
  int ho = (idx / WP) % HP;
  int c  = (idx / (size_t)HWSZ) % 32;
  int b  = idx / ((size_t)HWSZ * 32);
  const float* p = x + ((size_t)(b * 32 + c) * 256 + 2 * ho) * 256 + 2 * wo;
  float m = fmaxf(fmaxf(p[0], p[1]), fmaxf(p[256], p[257]));
  xp[idx] = (bf16_t)m;
}

// h0 fp32 -> bf16 into channels [64,128) of the concat buffer
__global__ __launch_bounds__(256) void pack_h0_kernel(
    const float* __restrict__ h0, bf16_t* __restrict__ cat) {
  size_t idx = (size_t)blockIdx.x * 256 + threadIdx.x;  // over 4*64*HW
  int hw = idx % HWSZ;
  int c  = (idx / HWSZ) % 64;
  int b  = idx / ((size_t)HWSZ * 64);
  cat[((size_t)(b * 128 + 64 + c)) * HWSZ + hw] = (bf16_t)h0[idx];
}

// --------------------------- batch-norm pieces -----------------------------

__global__ __launch_bounds__(256) void bn_stats_kernel(
    const float* __restrict__ y, float* __restrict__ mean,
    float* __restrict__ rstd, int C) {
  int c = blockIdx.x;
  float s = 0.f, s2 = 0.f;
  for (int b = 0; b < BATCH; ++b) {
    const float* p = y + (size_t)(b * C + c) * HWSZ;
    for (int i = threadIdx.x; i < HWSZ; i += 256) {
      float v = p[i];
      s += v;
      s2 += v * v;
    }
  }
  __shared__ float sh[256], sh2[256];
  sh[threadIdx.x] = s;
  sh2[threadIdx.x] = s2;
  __syncthreads();
  for (int st = 128; st > 0; st >>= 1) {
    if (threadIdx.x < st) {
      sh[threadIdx.x] += sh[threadIdx.x + st];
      sh2[threadIdx.x] += sh2[threadIdx.x + st];
    }
    __syncthreads();
  }
  if (threadIdx.x == 0) {
    const float inv = 1.f / (float)(BATCH * HWSZ);
    float m = sh[0] * inv;
    float var = sh2[0] * inv - m * m;
    mean[c] = m;
    rstd[c] = rsqrtf(var + 1e-5f);
  }
}

// normalize + scale/shift + relu + pack to bf16 at channel stride dstC
__global__ __launch_bounds__(256) void bn_relu_pack_kernel(
    const float* __restrict__ y, const float* __restrict__ mean,
    const float* __restrict__ rstd, const float* __restrict__ gamma,
    const float* __restrict__ beta, bf16_t* __restrict__ dst, int C, int dstC) {
  size_t idx = (size_t)blockIdx.x * 256 + threadIdx.x;  // over B*C*HW
  int hw = idx % HWSZ;
  int c  = (idx / HWSZ) % C;
  int b  = idx / ((size_t)HWSZ * C);
  float v = (y[idx] - mean[c]) * rstd[c] * gamma[c] + beta[c];
  v = v > 0.f ? v : 0.f;
  dst[((size_t)(b * dstC + c)) * HWSZ + hw] = (bf16_t)v;
}

// --------------------------- WMMA fragment utils ---------------------------

__device__ __forceinline__ bfx16 load_frag_a(const bf16_t* Alds, int row, int grp) {
  const bf16_t* p = Alds + row * KC;
  bfx8 lo = *(const bfx8*)(p + 8 * grp);        // K = 8*grp .. +7
  bfx8 hi = *(const bfx8*)(p + 16 + 8 * grp);   // K = 16+8*grp .. +7
  bfx16 r;
#pragma unroll
  for (int i = 0; i < 8; ++i) {
    r[i] = lo[i];
    r[8 + i] = hi[i];
  }
  return r;
}

__device__ __forceinline__ bfx16 load_frag_b(const bf16_t* Blds, int pix, int grp) {
  // lane holds K = 8*grp .. +7 and 16+8*grp .. +7 for its pixel column
  const bf16_t* p = Blds + pix * KC;
  bfx8 lo = *(const bfx8*)(p + 8 * grp);
  bfx8 hi = *(const bfx8*)(p + 16 + 8 * grp);
  bfx16 r;
#pragma unroll
  for (int i = 0; i < 8; ++i) {
    r[i] = lo[i];
    r[8 + i] = hi[i];
  }
  return r;
}

__device__ __forceinline__ fx8 wmma_bf16(bfx16 a, bfx16 b, fx8 c) {
  return __builtin_amdgcn_wmma_f32_16x16x32_bf16(false, a, false, b, (short)0,
                                                 c, false, false);
}

// ------------------- 3x3 SAME conv, implicit GEMM, WMMA --------------------
// Workgroup: 64 out-channels x 64 pixels (one row strip). 8 waves; each wave
// owns one 16-row M-subtile and two 16-pixel N-subtiles (A-frag reuse).
// K = CIN*9 streamed in chunks of 32; weights staged by the TDM into a
// ping-pong LDS buffer (software pipeline, one barrier per K-step).

template <int CIN, int COUT>
__global__ __launch_bounds__(256) void conv3x3_wmma_kernel(
    const bf16_t* __restrict__ in, const bf16_t* __restrict__ wt,
    const float* __restrict__ bias, float* __restrict__ out) {
  constexpr int KTOT = CIN * 9;
  constexpr int NIT = KTOT / KC;
  __shared__ __attribute__((aligned(32))) bf16_t Alds[2][64 * KC];
  __shared__ __attribute__((aligned(32))) bf16_t Blds[2][TILE_PIX * KC];

  const int strip = blockIdx.x;
  const int w0 = (strip & 1) * 64;
  const int h  = (strip >> 1) & 127;
  const int b  = strip >> 8;
  const int obase = blockIdx.y * 64;

  const int tid  = threadIdx.x;
  const int lane = tid & 31;
  const int wid  = tid >> 5;
  const int msub = wid >> 1;        // 0..3
  const int nsb  = (wid & 1) * 2;   // 0 or 2 (two N-subtiles per wave)
  const int col  = lane & 15;
  const int grp  = lane >> 4;

  const int bkk   = tid >> 3;        // 0..31
  const int bpix0 = (tid & 7) * 8;   // 0..56

  auto stage_b = [&](int k0, bf16_t* Bdst) {
    int k = k0 + bkk;
    int ci = k / 9;
    int t = k - ci * 9;
    int ky = t / 3 - 1;
    int kx = t - (t / 3) * 3 - 1;
    int hh = h + ky;
#pragma unroll
    for (int j = 0; j < 8; ++j) {
      int pix = bpix0 + j;
      int ww = w0 + pix + kx;
      bf16_t v = (bf16_t)0.f;
      if ((unsigned)hh < (unsigned)HP && (unsigned)ww < (unsigned)WP)
        v = in[((size_t)(b * CIN + ci) * HP + hh) * WP + ww];
      Bdst[pix * KC + bkk] = v;
    }
  };
  auto stage_a = [&](int k0, int buf) {
    tdm_load_2d_bf16(wt + (size_t)obase * KTOT + k0,
                     (unsigned)(__UINTPTR_TYPE__)(void*)Alds[buf], KC, 64,
                     (unsigned)(COUT - obase), (unsigned)KTOT);
  };

  fx8 acc0 = {}, acc1 = {};

  // pipeline prologue
  if (tid < 32) stage_a(0, 0);
  stage_b(0, Blds[0]);
  if (tid < 32) tdm_wait();
  __syncthreads();

  for (int i = 0; i < NIT; ++i) {
    const int cur = i & 1;
    if (i + 1 < NIT) {
      if (tid < 32) stage_a((i + 1) * KC, 1 - cur);  // async DMA next A tile
      stage_b((i + 1) * KC, Blds[1 - cur]);          // loads hide under WMMA
    }
    bfx16 afrag = load_frag_a(Alds[cur], msub * 16 + col, grp);
    bfx16 b0 = load_frag_b(Blds[cur], (nsb + 0) * 16 + col, grp);
    bfx16 b1 = load_frag_b(Blds[cur], (nsb + 1) * 16 + col, grp);
    acc0 = wmma_bf16(afrag, b0, acc0);
    acc1 = wmma_bf16(afrag, b1, acc1);
    if (i + 1 < NIT) {
      if (tid < 32) tdm_wait();
      __syncthreads();
    }
  }

#pragma unroll
  for (int r = 0; r < 8; ++r) {
    int o = obase + msub * 16 + r + 8 * grp;
    if (o < COUT) {
      size_t rowb = ((size_t)(b * COUT + o) * HP + h) * WP;
      float bi = bias[o];
      out[rowb + w0 + (nsb + 0) * 16 + col] = acc0[r] + bi;
      out[rowb + w0 + (nsb + 1) * 16 + col] = acc1[r] + bi;
    }
  }
}

// ------------- deformable sampling einsum as WMMA implicit GEMM ------------
// y[b,o,h,w] = sum_{c,n} bilinear(act[b,c], (h,w)+tap_n+off) * w[o, c*9+n]
// COUT = 64. Tap tables (9 taps x 64 pixels) precomputed in LDS.

template <int CIN>
__global__ __launch_bounds__(256) void deform_wmma_kernel(
    const bf16_t* __restrict__ act, const float* __restrict__ off,
    const bf16_t* __restrict__ wt, float* __restrict__ out) {
  constexpr int KTOT = CIN * 9;
  constexpr int NIT = KTOT / KC;
  constexpr int COUT = 64;
  __shared__ __attribute__((aligned(32))) bf16_t Alds[2][64 * KC];
  __shared__ __attribute__((aligned(32))) bf16_t Blds[2][TILE_PIX * KC];
  __shared__ int   ty0[9 * TILE_PIX];
  __shared__ int   tx0[9 * TILE_PIX];
  __shared__ float tfy[9 * TILE_PIX];
  __shared__ float tfx[9 * TILE_PIX];

  const int strip = blockIdx.x;
  const int w0 = (strip & 1) * 64;
  const int h  = (strip >> 1) & 127;
  const int b  = strip >> 8;

  const int tid  = threadIdx.x;
  const int lane = tid & 31;
  const int wid  = tid >> 5;
  const int msub = wid >> 1;
  const int nsb  = (wid & 1) * 2;
  const int col  = lane & 15;
  const int grp  = lane >> 4;

  const int bkk   = tid >> 3;
  const int bpix0 = (tid & 7) * 8;

  // precompute per-(tap,pixel) sampling data; off has 18 channels
  // (dy = ch 0..8, dx = ch 9..17)
  for (int t = tid; t < 9 * TILE_PIX; t += 256) {
    int pix = t & 63;
    int n = t >> 6;
    int w = w0 + pix;
    float dy = off[((size_t)(b * 18 + n) * HP + h) * WP + w];
    float dx = off[((size_t)(b * 18 + 9 + n) * HP + h) * WP + w];
    float sy = (float)h + (float)(n / 3 - 1) + dy;
    float sx = (float)w + (float)(n % 3 - 1) + dx;
    float fy = floorf(sy), fx = floorf(sx);
    ty0[t] = (int)fy;
    tx0[t] = (int)fx;
    tfy[t] = sy - fy;
    tfx[t] = sx - fx;
  }
  __syncthreads();

  auto stage_b = [&](int k0, bf16_t* Bdst) {
    int k = k0 + bkk;
    int ci = k / 9;
    int n = k - ci * 9;
    const bf16_t* plane = act + (size_t)(b * CIN + ci) * HWSZ;
#pragma unroll
    for (int j = 0; j < 8; ++j) {
      int pix = bpix0 + j;
      int t = (n << 6) + pix;
      int y0 = ty0[t], x0 = tx0[t];
      float fy = tfy[t], fx = tfx[t];
      float v00 = ((unsigned)y0 < (unsigned)HP && (unsigned)x0 < (unsigned)WP)
                      ? (float)plane[y0 * WP + x0] : 0.f;
      float v01 = ((unsigned)y0 < (unsigned)HP && (unsigned)(x0 + 1) < (unsigned)WP)
                      ? (float)plane[y0 * WP + x0 + 1] : 0.f;
      float v10 = ((unsigned)(y0 + 1) < (unsigned)HP && (unsigned)x0 < (unsigned)WP)
                      ? (float)plane[(y0 + 1) * WP + x0] : 0.f;
      float v11 = ((unsigned)(y0 + 1) < (unsigned)HP && (unsigned)(x0 + 1) < (unsigned)WP)
                      ? (float)plane[(y0 + 1) * WP + x0 + 1] : 0.f;
      float v = (1.f - fy) * ((1.f - fx) * v00 + fx * v01) +
                fy * ((1.f - fx) * v10 + fx * v11);
      Bdst[pix * KC + bkk] = (bf16_t)v;
    }
  };
  auto stage_a = [&](int k0, int buf) {
    tdm_load_2d_bf16(wt + (size_t)k0,
                     (unsigned)(__UINTPTR_TYPE__)(void*)Alds[buf], KC, 64,
                     (unsigned)COUT, (unsigned)KTOT);
  };

  fx8 acc0 = {}, acc1 = {};

  if (tid < 32) stage_a(0, 0);
  stage_b(0, Blds[0]);
  if (tid < 32) tdm_wait();
  __syncthreads();

  for (int i = 0; i < NIT; ++i) {
    const int cur = i & 1;
    if (i + 1 < NIT) {
      if (tid < 32) stage_a((i + 1) * KC, 1 - cur);
      stage_b((i + 1) * KC, Blds[1 - cur]);
    }
    bfx16 afrag = load_frag_a(Alds[cur], msub * 16 + col, grp);
    bfx16 b0 = load_frag_b(Blds[cur], (nsb + 0) * 16 + col, grp);
    bfx16 b1 = load_frag_b(Blds[cur], (nsb + 1) * 16 + col, grp);
    acc0 = wmma_bf16(afrag, b0, acc0);
    acc1 = wmma_bf16(afrag, b1, acc1);
    if (i + 1 < NIT) {
      if (tid < 32) tdm_wait();
      __syncthreads();
    }
  }

#pragma unroll
  for (int r = 0; r < 8; ++r) {
    int o = msub * 16 + r + 8 * grp;
    size_t rowb = ((size_t)(b * COUT + o) * HP + h) * WP;
    out[rowb + w0 + (nsb + 0) * 16 + col] = acc0[r];
    out[rowb + w0 + (nsb + 1) * 16 + col] = acc1[r];
  }
}

// ---------------- gates conv (256x1152) fused with LSTM cell ---------------
// Workgroup: all 256 out-channels x 64 pixels, so i/f/o/g for every pixel
// are resident; LSTM runs in the epilogue via LDS and h,c go straight to
// d_out. 8 waves x (4 M-subtiles x 2 N-subtiles) = 8 WMMAs per wave/K-step.

__device__ __forceinline__ float sigmoidf_(float v) {
  return 1.f / (1.f + __expf(-v));
}

__global__ __launch_bounds__(256) void gates_lstm_wmma_kernel(
    const bf16_t* __restrict__ in,   // concat (B,128,H,W) bf16
    const bf16_t* __restrict__ wt,   // (256,1152) bf16
    const float* __restrict__ bias,  // 256
    const float* __restrict__ c0,    // (B,64,H,W) fp32
    float* __restrict__ out) {       // h then c, each (B,64,H,W)
  constexpr int CIN = 128, COUT = 256;
  constexpr int KTOT = CIN * 9;
  constexpr int NIT = KTOT / KC;
  // 64 KB LDS: ping-pong A (2x16KB) + ping-pong B (2x4KB) while looping,
  // then reused as the 256x64 fp32 gate staging block for the LSTM epilogue.
  __shared__ __attribute__((aligned(32))) char smem[COUT * TILE_PIX * 4];
  bf16_t* AldsBase = (bf16_t*)smem;                       // [2][256*KC]
  bf16_t* BldsBase = (bf16_t*)(smem + 2 * COUT * KC * 2); // [2][64*KC]
  float*  Gres = (float*)smem;                            // [256*64]

  const int strip = blockIdx.x;
  const int w0 = (strip & 1) * 64;
  const int h  = (strip >> 1) & 127;
  const int b  = strip >> 8;

  const int tid  = threadIdx.x;
  const int lane = tid & 31;
  const int wid  = tid >> 5;
  const int mq   = wid >> 1;        // 0..3 -> out-channel rows mq*64..+63
  const int nsb  = (wid & 1) * 2;
  const int col  = lane & 15;
  const int grp  = lane >> 4;

  const int bkk   = tid >> 3;
  const int bpix0 = (tid & 7) * 8;

  auto stage_b = [&](int k0, bf16_t* Bdst) {
    int k = k0 + bkk;
    int ci = k / 9;
    int t = k - ci * 9;
    int ky = t / 3 - 1;
    int kx = t - (t / 3) * 3 - 1;
    int hh = h + ky;
#pragma unroll
    for (int j = 0; j < 8; ++j) {
      int pix = bpix0 + j;
      int ww = w0 + pix + kx;
      bf16_t v = (bf16_t)0.f;
      if ((unsigned)hh < (unsigned)HP && (unsigned)ww < (unsigned)WP)
        v = in[((size_t)(b * CIN + ci) * HP + hh) * WP + ww];
      Bdst[pix * KC + bkk] = v;
    }
  };
  auto stage_a = [&](int k0, int buf) {
    tdm_load_2d_bf16(wt + (size_t)k0,
                     (unsigned)(__UINTPTR_TYPE__)(void*)(AldsBase + buf * COUT * KC),
                     KC, COUT, (unsigned)COUT, (unsigned)KTOT);
  };

  fx8 acc[4][2] = {};

  if (tid < 32) stage_a(0, 0);
  stage_b(0, BldsBase);
  if (tid < 32) tdm_wait();
  __syncthreads();

  for (int i = 0; i < NIT; ++i) {
    const int cur = i & 1;
    const bf16_t* Ab = AldsBase + cur * COUT * KC;
    const bf16_t* Bb = BldsBase + cur * TILE_PIX * KC;
    if (i + 1 < NIT) {
      if (tid < 32) stage_a((i + 1) * KC, 1 - cur);
      stage_b((i + 1) * KC, BldsBase + (1 - cur) * TILE_PIX * KC);
    }
    bfx16 b0 = load_frag_b(Bb, (nsb + 0) * 16 + col, grp);
    bfx16 b1 = load_frag_b(Bb, (nsb + 1) * 16 + col, grp);
#pragma unroll
    for (int ms = 0; ms < 4; ++ms) {
      bfx16 af = load_frag_a(Ab, (mq * 4 + ms) * 16 + col, grp);
      acc[ms][0] = wmma_bf16(af, b0, acc[ms][0]);
      acc[ms][1] = wmma_bf16(af, b1, acc[ms][1]);
    }
    if (i + 1 < NIT) {
      if (tid < 32) tdm_wait();
      __syncthreads();
    }
  }

  // epilogue: stage gates (+bias) into LDS, then LSTM pointwise
  __syncthreads();  // all LDS reads of A/B done before reuse as Gres
#pragma unroll
  for (int ms = 0; ms < 4; ++ms) {
#pragma unroll
    for (int t = 0; t < 2; ++t) {
#pragma unroll
      for (int r = 0; r < 8; ++r) {
        int o = (mq * 4 + ms) * 16 + r + 8 * grp;
        int pix = (nsb + t) * 16 + col;
        Gres[o * TILE_PIX + pix] = acc[ms][t][r] + bias[o];
      }
    }
  }
  __syncthreads();

  for (int t = tid; t < 64 * TILE_PIX; t += 256) {
    int c = t >> 6;      // 0..63
    int pix = t & 63;
    float ig = Gres[(c)       * TILE_PIX + pix];
    float fg = Gres[(64 + c)  * TILE_PIX + pix];
    float og = Gres[(128 + c) * TILE_PIX + pix];
    float gg = Gres[(192 + c) * TILE_PIX + pix];
    size_t idx = ((size_t)(b * 64 + c) * HP + h) * WP + w0 + pix;
    float cn = sigmoidf_(fg) * c0[idx] + sigmoidf_(ig) * tanhf(gg);
    float hn = sigmoidf_(og) * tanhf(cn);
    out[idx] = hn;                               // h
    out[(size_t)BATCH * 64 * HWSZ + idx] = cn;   // c
  }
}

// ------------------------------- launcher ----------------------------------

extern "C" void kernel_launch(void* const* d_in, const int* in_sizes, int n_in,
                              void* d_out, int out_size, void* d_ws,
                              size_t ws_size, hipStream_t stream) {
  const float* x       = (const float*)d_in[0];
  const float* h0      = (const float*)d_in[1];
  const float* c0      = (const float*)d_in[2];
  const float* w_off1  = (const float*)d_in[3];
  const float* b_off1  = (const float*)d_in[4];
  const float* w1      = (const float*)d_in[5];
  const float* g1      = (const float*)d_in[6];
  const float* be1     = (const float*)d_in[7];
  const float* w_off2  = (const float*)d_in[8];
  const float* b_off2  = (const float*)d_in[9];
  const float* w2      = (const float*)d_in[10];
  const float* g2      = (const float*)d_in[11];
  const float* be2     = (const float*)d_in[12];
  const float* w_gates = (const float*)d_in[13];
  const float* b_gates = (const float*)d_in[14];
  float* out = (float*)d_out;

  char* base = (char*)d_ws;
  size_t cur = 0;
  auto alloc = [&](size_t bytes) -> void* {
    cur = (cur + 255) & ~(size_t)255;
    void* p = base + cur;
    cur += bytes;
    return p;
  };

  const size_t nxp = (size_t)BATCH * 32 * HWSZ;        // 2,097,152
  const size_t nact = (size_t)BATCH * 64 * HWSZ;       // 4,194,304
  bf16_t* xp     = (bf16_t*)alloc(nxp * 2);
  bf16_t* wboff1 = (bf16_t*)alloc((size_t)18 * 288 * 2);
  bf16_t* w1bf   = (bf16_t*)alloc((size_t)64 * 288 * 2);
  bf16_t* wboff2 = (bf16_t*)alloc((size_t)18 * 576 * 2);
  bf16_t* w2bf   = (bf16_t*)alloc((size_t)64 * 576 * 2);
  bf16_t* wgbf   = (bf16_t*)alloc((size_t)256 * 1152 * 2);
  float*  offb   = (float*)alloc((size_t)BATCH * 18 * HWSZ * 4);
  float*  ybuf   = (float*)alloc(nact * 4);
  bf16_t* a1bf   = (bf16_t*)alloc(nact * 2);
  bf16_t* catbf  = (bf16_t*)alloc((size_t)BATCH * 128 * HWSZ * 2);
  float*  mean1  = (float*)alloc(64 * 4);
  float*  rstd1  = (float*)alloc(64 * 4);
  float*  mean2  = (float*)alloc(64 * 4);
  float*  rstd2  = (float*)alloc(64 * 4);

  auto blocks = [](size_t n) { return (unsigned)((n + 255) / 256); };

  // weight packs (fp32 -> bf16, OIHW flat == (O, C*9+tap) GEMM layout)
  pack_bf16_kernel<<<blocks(18 * 288), 256, 0, stream>>>(w_off1, wboff1, 18 * 288);
  pack_bf16_kernel<<<blocks(64 * 288), 256, 0, stream>>>(w1, w1bf, 64 * 288);
  pack_bf16_kernel<<<blocks(18 * 576), 256, 0, stream>>>(w_off2, wboff2, 18 * 576);
  pack_bf16_kernel<<<blocks(64 * 576), 256, 0, stream>>>(w2, w2bf, 64 * 576);
  pack_bf16_kernel<<<blocks(256 * 1152), 256, 0, stream>>>(w_gates, wgbf, 256 * 1152);

  // stage 0: maxpool (fp32 -> bf16)
  maxpool_bf_kernel<<<blocks(nxp), 256, 0, stream>>>(x, xp);

  const dim3 gstrip(1024, 1), wg(256);

  // layer 1: offsets conv, deformable einsum, BN+ReLU -> a1 (bf16)
  conv3x3_wmma_kernel<32, 18><<<gstrip, wg, 0, stream>>>(xp, wboff1, b_off1, offb);
  deform_wmma_kernel<32><<<1024, wg, 0, stream>>>(xp, offb, w1bf, ybuf);
  bn_stats_kernel<<<64, 256, 0, stream>>>(ybuf, mean1, rstd1, 64);
  bn_relu_pack_kernel<<<blocks(nact), 256, 0, stream>>>(ybuf, mean1, rstd1, g1,
                                                        be1, a1bf, 64, 64);

  // layer 2: same, BN+ReLU packs into concat buffer channels [0,64)
  conv3x3_wmma_kernel<64, 18><<<gstrip, wg, 0, stream>>>(a1bf, wboff2, b_off2, offb);
  deform_wmma_kernel<64><<<1024, wg, 0, stream>>>(a1bf, offb, w2bf, ybuf);
  bn_stats_kernel<<<64, 256, 0, stream>>>(ybuf, mean2, rstd2, 64);
  bn_relu_pack_kernel<<<blocks(nact), 256, 0, stream>>>(ybuf, mean2, rstd2, g2,
                                                        be2, catbf, 64, 128);
  pack_h0_kernel<<<blocks(nact), 256, 0, stream>>>(h0, catbf);

  // fused gates conv + LSTM: h,c straight to d_out, no gates tensor in HBM
  gates_lstm_wmma_kernel<<<1024, wg, 0, stream>>>(catbf, wgbf, b_gates, c0, out);

  (void)in_sizes; (void)n_in; (void)out_size; (void)ws_size;
}
